// SensorWiseMHA_43379169690057
// MI455X (gfx1250) — matrix operand order repaired and checked by reading, not compiled, not run on hardware
//
#include <hip/hip_runtime.h>
#include <hip/hip_bf16.h>
#include <stdint.h>

// ---------- types ----------
typedef _Float16 h16;
typedef __attribute__((ext_vector_type(16))) _Float16 h16x16;
typedef __attribute__((ext_vector_type(8)))  _Float16 h16x8;
typedef __attribute__((ext_vector_type(8)))  float    f32x8;

#define B_   2
#define L_   2048
#define D_   512
#define H_   8
#define HD_  64
#define M_   (B_ * L_)   // 4096
#define N3_  (3 * D_)    // 1536

static __device__ __forceinline__ f32x8 wmma16(h16x16 a, h16x16 b, f32x8 c) {
  // D = A(16x32 f16) * B(32x16 f16) + C(16x16 f32)
  return __builtin_amdgcn_wmma_f32_16x16x32_f16(false, a, false, b, (short)0, c,
                                                false, false);
}

static __device__ __forceinline__ h16x16 cat8(h16x8 lo, h16x8 hi) {
  return __builtin_shufflevector(lo, hi, 0,1,2,3,4,5,6,7,8,9,10,11,12,13,14,15);
}

// A-fragment (16x32, K-dim=32) from row-major [M][stride] f16.
// ISA layout: lanes 0-15 hold M=lane, K={(lane/16)*8 .. +7} U {16+(lane/16)*8 .. +7}
static __device__ __forceinline__ h16x16 load_frag_a(const h16* base, int stride, int lane) {
  const int m  = lane & 15;
  const int kc = (lane >> 4) << 3;                 // 0 or 8
  const h16* p = base + (size_t)m * stride + kc;
  h16x8 lo = *(const h16x8*)(p);
  h16x8 hi = *(const h16x8*)(p + 16);
  return cat8(lo, hi);
}

// B-fragment (32x16, K-dim=32) from B^T stored row-major [N][stride].
// ISA layout: lane holds N=lane%16, K={(lane/16)*8 .. +7} U {16+(lane/16)*8 .. +7}
static __device__ __forceinline__ h16x16 load_frag_b(const h16* base, int stride, int lane) {
  const int n  = lane & 15;
  const int kc = (lane >> 4) << 3;                 // 0 or 8
  const h16* p = base + (size_t)n * stride + kc;
  h16x8 lo = *(const h16x8*)(p);
  h16x8 hi = *(const h16x8*)(p + 16);
  return cat8(lo, hi);
}

// ---------- precision-cast kernels ----------
__global__ void __launch_bounds__(256) cast_f32_to_f16(const float* __restrict__ in,
                                                       h16* __restrict__ out, int n) {
  int i = blockIdx.x * 256 + threadIdx.x;
  if (i < n) out[i] = (h16)in[i];
}

// in: [K][N] f32  ->  out: [N][K] f16
__global__ void __launch_bounds__(256) cast_transpose(const float* __restrict__ in,
                                                      h16* __restrict__ out,
                                                      int K, int N) {
  int i = blockIdx.x * 256 + threadIdx.x;
  if (i < K * N) {
    int n = i / K, k = i - n * K;
    out[(size_t)n * K + k] = (h16)in[(size_t)k * N + n];
  }
}

// ---------- QKV projection: [4096x512] x [512x1536] ----------
// Wave tile 64(M) x 32(N): 8 WMMA per K-step, A-frags reused 2x, B-frags 4x.
__global__ void __launch_bounds__(128)
qkv_gemm(const h16* __restrict__ A,    // x f16 [M][512]
         const h16* __restrict__ BT,   // W_qkv^T f16 [1536][512]
         const float* __restrict__ bias,
         h16* __restrict__ Q,          // [B,H,L,HD]  (pre-scaled by 1/sqrt(HD))
         h16* __restrict__ Kt,         // [B,H,L,HD]
         h16* __restrict__ Vt) {       // [B,H,HD,L]  (transposed for P@V)
  const int lane = threadIdx.x & 31;
  const int wave = threadIdx.x >> 5;
  const int wm = wave >> 1, wn = wave & 1;
  const int m0 = blockIdx.x * 128 + wm * 64;
  const int n0 = blockIdx.y * 64 + wn * 32;

  f32x8 acc[4][2] = {};
  #pragma unroll 4
  for (int kk = 0; kk < D_; kk += 32) {
    h16x16 af[4], bf[2];
    #pragma unroll
    for (int mt = 0; mt < 4; ++mt)
      af[mt] = load_frag_a(A + (size_t)(m0 + mt * 16) * D_ + kk, D_, lane);
    #pragma unroll
    for (int nt = 0; nt < 2; ++nt)
      bf[nt] = load_frag_b(BT + (size_t)(n0 + nt * 16) * D_ + kk, D_, lane);
    #pragma unroll
    for (int mt = 0; mt < 4; ++mt)
      #pragma unroll
      for (int nt = 0; nt < 2; ++nt)
        acc[mt][nt] = wmma16(af[mt], bf[nt], acc[mt][nt]);
  }

  // Epilogue: which/h/b are wave-uniform (tiles never straddle 512/64/2048
  // boundaries) -> force to SGPRs and branch once per tile, not per element.
  #pragma unroll
  for (int mt = 0; mt < 4; ++mt) {
    const int mrow0 = m0 + mt * 16 + ((lane >> 4) << 3);   // + i
    const int b  = __builtin_amdgcn_readfirstlane(mrow0 >> 11);
    const int l0 = mrow0 & (L_ - 1);
    #pragma unroll
    for (int nt = 0; nt < 2; ++nt) {
      const int nbase = n0 + nt * 16;                      // wave-uniform
      const int which = __builtin_amdgcn_readfirstlane(nbase >> 9);   // 0=Q 1=K 2=V
      const int h     = __builtin_amdgcn_readfirstlane((nbase >> 6) & 7);
      const int d     = (nbase & 63) + (lane & 15);
      const float bv  = bias[nbase + (lane & 15)];
      const size_t bh = (size_t)(b * H_ + h);
      if (which == 0) {
        h16* dst = Q + (bh * L_ + l0) * HD_ + d;
        #pragma unroll
        for (int i = 0; i < 8; ++i)
          dst[(size_t)i * HD_] = (h16)((acc[mt][nt][i] + bv) * 0.125f);
      } else if (which == 1) {
        h16* dst = Kt + (bh * L_ + l0) * HD_ + d;
        #pragma unroll
        for (int i = 0; i < 8; ++i)
          dst[(size_t)i * HD_] = (h16)(acc[mt][nt][i] + bv);
      } else {
        h16* dst = Vt + (bh * HD_ + d) * L_ + l0;          // contiguous in l
        #pragma unroll
        for (int i = 0; i < 8; ++i)
          dst[i] = (h16)(acc[mt][nt][i] + bv);
      }
    }
  }
}

// ---------- Flash attention: per-wave 16-row q tile, stream 32 keys ----------
__global__ void __launch_bounds__(128)
flash_attn(const h16* __restrict__ Q, const h16* __restrict__ Kb,
           const h16* __restrict__ Vt, h16* __restrict__ O /* [B*L][512] */) {
  const int lane = threadIdx.x & 31;
  const int wave = threadIdx.x >> 5;
  const int bh = blockIdx.y;             // 0..15
  const int b = bh >> 3, h = bh & 7;
  const int q0 = blockIdx.x * 64 + wave * 16;

  const h16* Qh = Q  + (size_t)bh * L_ * HD_;
  const h16* Kh = Kb + (size_t)bh * L_ * HD_;
  const h16* Vh = Vt + (size_t)bh * HD_ * L_;

  // per-wave private LDS slab: no cross-wave sharing -> no block barriers needed
  __shared__ __align__(16) h16 plds[4][16][32];

  const h16x16 qa0 = load_frag_a(Qh + (size_t)q0 * HD_ + 0,  HD_, lane);
  const h16x16 qa1 = load_frag_a(Qh + (size_t)q0 * HD_ + 32, HD_, lane);

  f32x8 o[4] = {};
  float mrun[8], lrun[8];
  #pragma unroll
  for (int i = 0; i < 8; ++i) { mrun[i] = -1e30f; lrun[i] = 0.f; }

  // 2x unroll: next iteration's K loads + score WMMAs overlap this
  // iteration's softmax dependency chain (shuffles + v_exp_f32).
  #pragma unroll 2
  for (int kb = 0; kb < L_; kb += 32) {
    // scores: 16 q-rows x 32 keys, K-dim = HD = 64 (chained WMMAs)
    // (softmax scale already folded into Q)
    f32x8 s[2];
    #pragma unroll
    for (int t = 0; t < 2; ++t) {
      h16x16 kf0 = load_frag_b(Kh + (size_t)(kb + t * 16) * HD_ + 0,  HD_, lane);
      h16x16 kf1 = load_frag_b(Kh + (size_t)(kb + t * 16) * HD_ + 32, HD_, lane);
      f32x8 z = {};
      z    = wmma16(qa0, kf0, z);
      s[t] = wmma16(qa1, kf1, z);
    }

    // online softmax (rows split across 16-lane halves -> width-16 shuffles)
    #pragma unroll
    for (int i = 0; i < 8; ++i) {
      float s0 = s[0][i], s1 = s[1][i];
      float mx = fmaxf(s0, s1);
      #pragma unroll
      for (int off = 8; off; off >>= 1) mx = fmaxf(mx, __shfl_xor(mx, off, 16));
      const float mnew  = fmaxf(mrun[i], mx);
      const float alpha = __expf(mrun[i] - mnew);
      const float p0 = __expf(s0 - mnew), p1 = __expf(s1 - mnew);
      float ps = p0 + p1;
      #pragma unroll
      for (int off = 8; off; off >>= 1) ps += __shfl_xor(ps, off, 16);
      lrun[i] = lrun[i] * alpha + ps;
      mrun[i] = mnew;
      #pragma unroll
      for (int t = 0; t < 4; ++t) o[t][i] *= alpha;
      const int row = i + ((lane >> 4) << 3);
      plds[wave][row][(lane & 15)]      = (h16)p0;
      plds[wave][row][16 + (lane & 15)] = (h16)p1;
    }
    // Same-wave LDS ops execute in order at the LDS unit; we only need to stop
    // the compiler from reordering the fragment load across the stores.
    __builtin_amdgcn_wave_barrier();
    const h16x16 pa = load_frag_a(&plds[wave][0][0], 32, lane);
    __builtin_amdgcn_wave_barrier();

    // O += P @ V  (V stored transposed [d][l] => contiguous B-frag loads)
    #pragma unroll
    for (int t = 0; t < 4; ++t) {
      h16x16 vf = load_frag_b(Vh + (size_t)(t * 16) * L_ + kb, L_, lane);
      o[t] = wmma16(pa, vf, o[t]);
    }
  }

  // epilogue: normalize and scatter into [B*L][512] row-major (head-concat)
  #pragma unroll
  for (int t = 0; t < 4; ++t) {
    const int col = h * 64 + t * 16 + (lane & 15);
    #pragma unroll
    for (int i = 0; i < 8; ++i) {
      const int q = q0 + i + ((lane >> 4) << 3);
      const float v = o[t][i] / lrun[i];
      O[((size_t)(b * L_ + q)) * D_ + col] = (h16)v;
    }
  }
}

// ---------- Output projection: [4096x512] x [512x512] -> fp32 out ----------
__global__ void __launch_bounds__(128)
out_gemm(const h16* __restrict__ A,    // attn out f16 [M][512]
         const h16* __restrict__ BT,   // W_out^T f16 [512][512]
         const float* __restrict__ bias,
         float* __restrict__ C) {      // [M][512] fp32
  const int lane = threadIdx.x & 31;
  const int wave = threadIdx.x >> 5;
  const int wm = wave >> 1, wn = wave & 1;
  const int m0 = blockIdx.x * 128 + wm * 64;
  const int n0 = blockIdx.y * 64 + wn * 32;

  f32x8 acc[4][2] = {};
  #pragma unroll 4
  for (int kk = 0; kk < D_; kk += 32) {
    h16x16 af[4], bf[2];
    #pragma unroll
    for (int mt = 0; mt < 4; ++mt)
      af[mt] = load_frag_a(A + (size_t)(m0 + mt * 16) * D_ + kk, D_, lane);
    #pragma unroll
    for (int nt = 0; nt < 2; ++nt)
      bf[nt] = load_frag_b(BT + (size_t)(n0 + nt * 16) * D_ + kk, D_, lane);
    #pragma unroll
    for (int mt = 0; mt < 4; ++mt)
      #pragma unroll
      for (int nt = 0; nt < 2; ++nt)
        acc[mt][nt] = wmma16(af[mt], bf[nt], acc[mt][nt]);
  }

  #pragma unroll
  for (int mt = 0; mt < 4; ++mt)
    #pragma unroll
    for (int nt = 0; nt < 2; ++nt) {
      const int n = n0 + nt * 16 + (lane & 15);
      const float bv = bias[n];
      float* dst = C + (size_t)(m0 + mt * 16 + ((lane >> 4) << 3)) * D_ + n;
      #pragma unroll
      for (int i = 0; i < 8; ++i)
        dst[(size_t)i * D_] = acc[mt][nt][i] + bv;
    }
}

// ---------- launch ----------
extern "C" void kernel_launch(void* const* d_in, const int* in_sizes, int n_in,
                              void* d_out, int out_size, void* d_ws, size_t ws_size,
                              hipStream_t stream) {
  const float* x    = (const float*)d_in[0];
  const float* Wqkv = (const float*)d_in[1];
  const float* bqkv = (const float*)d_in[2];
  const float* Wout = (const float*)d_in[3];
  const float* bout = (const float*)d_in[4];
  float* out = (float*)d_out;

  uintptr_t p = (uintptr_t)d_ws;
  auto alloc = [&](size_t bytes) -> void* {
    p = (p + 255) & ~(uintptr_t)255;
    void* r = (void*)p;
    p += bytes;
    return r;
  };
  h16* xb    = (h16*)alloc((size_t)M_ * D_ * sizeof(h16));   // x f16
  h16* wqkvT = (h16*)alloc((size_t)N3_ * D_ * sizeof(h16));  // Wqkv^T f16
  h16* woutT = (h16*)alloc((size_t)D_ * D_ * sizeof(h16));   // Wout^T f16
  h16* Qb    = (h16*)alloc((size_t)M_ * D_ * sizeof(h16));
  h16* Kb    = (h16*)alloc((size_t)M_ * D_ * sizeof(h16));
  h16* Vt    = (h16*)alloc((size_t)M_ * D_ * sizeof(h16));
  h16* Ob    = (h16*)alloc((size_t)M_ * D_ * sizeof(h16));

  const int nx = M_ * D_;
  cast_f32_to_f16<<<(nx + 255) / 256, 256, 0, stream>>>(x, xb, nx);
  cast_transpose<<<(D_ * N3_ + 255) / 256, 256, 0, stream>>>(Wqkv, wqkvT, D_, N3_);
  cast_transpose<<<(D_ * D_ + 255) / 256, 256, 0, stream>>>(Wout, woutT, D_, D_);

  qkv_gemm<<<dim3(M_ / 128, N3_ / 64), 128, 0, stream>>>(xb, wqkvT, bqkv, Qb, Kb, Vt);
  flash_attn<<<dim3(L_ / 64, B_ * H_), 128, 0, stream>>>(Qb, Kb, Vt, Ob);
  out_gemm<<<dim3(M_ / 128, D_ / 64), 128, 0, stream>>>(Ob, woutT, bout, out);
}
